// MoEModel_83665962926118
// MI455X (gfx1250) — matrix operand order repaired and checked by reading, not compiled, not run on hardware
//
#include <hip/hip_runtime.h>
#include <hip/hip_bf16.h>

// MoE fused forward for MI455X (gfx1250, wave32, WMMA).
// Pipeline: pack weights fp32->f16 into WMMA B-fragment layout (d_ws),
// then one fused kernel: extractor -> 7 experts -> gate -> outputs.

typedef __attribute__((ext_vector_type(16))) _Float16 v16h;
typedef __attribute__((ext_vector_type(8)))  float    v8f;

union Frag {
    v16h  h;
    uint4 q[2];
};

// A-fragment (16x32 f16, M x K) load from row-major f16 LDS tile.
// ISA 7.12.2: lane m<16 holds row m, K {kb..kb+7} in v0..3 and {kb+16..kb+23}
// in v4..7 with kb = 0 (lanes 0-15) or 8 (lanes 16-31).
__device__ __forceinline__ Frag ldsA(const _Float16* p, int ld, int kt, int lane) {
    Frag a;
    int m  = lane & 15;
    int kb = kt * 32 + ((lane & 16) ? 8 : 0);
    a.q[0] = *(const uint4*)(p + m * ld + kb);
    a.q[1] = *(const uint4*)(p + m * ld + kb + 16);
    return a;
}

// B-fragment from pre-packed workspace: lane reads 8 consecutive dwords.
__device__ __forceinline__ Frag wsB(const unsigned* t, int lane) {
    Frag b;
    const uint4* p = (const uint4*)(t + lane * 8);
    b.q[0] = p[0];
    b.q[1] = p[1];
    return b;
}

__device__ __forceinline__ v8f wmma16(const Frag& a, const Frag& b, v8f c) {
    return __builtin_amdgcn_wmma_f32_16x16x32_f16(
        /*neg_a=*/false, a.h, /*neg_b=*/false, b.h,
        /*c_mod=*/(short)0, c, /*reuse_a=*/false, /*reuse_b=*/false);
}

// ---------------------------------------------------------------------------
// Weight packing: fp32 row-major [K x N] -> f16 WMMA B-fragment tiles.
// Tile (nt,kt) is 256 dwords: dword index = lane*8 + i, packing
// (k0,k0+1) for column n = nt*16 + (lane&15),
// k0 = kt*32 + (lane&16 ? 8 : 0) + 2*i + (i >= 4 ? 8 : 0).   Zero-pad columns n >= N.
// ---------------------------------------------------------------------------
__global__ void pack_b(const float* __restrict__ src, unsigned* __restrict__ dst,
                       int K, int N, int Npad, long sStride, long dStride, long total) {
    long tid = (long)blockIdx.x * blockDim.x + threadIdx.x;
    if (tid >= total) return;
    int KT = K >> 5, NT = Npad >> 4;
    int i    = (int)(tid & 7);
    int lane = (int)((tid >> 3) & 31);
    long t   = tid >> 8;
    int kt = (int)(t % KT); t /= KT;
    int nt = (int)(t % NT); t /= NT;
    int b  = (int)t;
    int n  = nt * 16 + (lane & 15);
    int k0 = kt * 32 + ((lane & 16) ? 8 : 0) + 2 * i + ((i >= 4) ? 8 : 0);
    const float* s = src + (long)b * sStride;
    float f0 = (n < N) ? s[(long)k0 * N + n] : 0.0f;
    float f1 = (n < N) ? s[(long)(k0 + 1) * N + n] : 0.0f;
    union { _Float16 h[2]; unsigned u; } pk;
    pk.h[0] = (_Float16)f0;
    pk.h[1] = (_Float16)f1;
    dst[(long)b * dStride + ((long)nt * KT + kt) * 256 + lane * 8 + i] = pk.u;
}

// ---------------------------------------------------------------------------
// Fused MoE forward. 128 threads = 4 wave32; each wave owns 16 rows.
// ---------------------------------------------------------------------------
__launch_bounds__(128)
__global__ void moe_fused(const float* __restrict__ x,
                          const int*   __restrict__ domain,
                          const float* __restrict__ b1,
                          const float* __restrict__ b2,
                          const float* __restrict__ br1,
                          const float* __restrict__ wr2,
                          const float* __restrict__ br2,
                          const float* __restrict__ bg,
                          const unsigned* __restrict__ W1F,
                          const unsigned* __restrict__ W2F,
                          const unsigned* __restrict__ Wr1F,
                          const unsigned* __restrict__ WgF,
                          float* __restrict__ outHard,
                          float* __restrict__ outSoft,
                          float* __restrict__ outW) {
    __shared__ __align__(16) _Float16 sh_h1[4][16 * 256];   // 32 KB
    __shared__ __align__(16) _Float16 sh_z [4][16 * 128];   // 16 KB
    __shared__ float sh_pred[4][16][8];                     // 2 KB

    const int lane = threadIdx.x & 31;
    const int wv   = threadIdx.x >> 5;
    const int n    = lane & 15;                 // C/D fragment column
    const int moff = (lane & 16) ? 8 : 0;       // C/D fragment row offset
    const long base = (long)blockIdx.x * 64 + wv * 16;

    _Float16* h1 = sh_h1[wv];
    _Float16* zz = sh_z[wv];

    // -------- stage 1: h1 = relu(x @ W1 + b1), M=16 K=64 N=256 ------------
    Frag ax[2];
    {
        int m  = lane & 15;
        int kb = (lane & 16) ? 8 : 0;
        const float* xr = x + (base + m) * 64 + kb;
        #pragma unroll
        for (int kt = 0; kt < 2; ++kt)
            #pragma unroll
            for (int j = 0; j < 8; ++j) {
                ax[kt].h[j]     = (_Float16)xr[kt * 32 + j];
                ax[kt].h[8 + j] = (_Float16)xr[kt * 32 + 16 + j];
            }
    }
    for (int nt = 0; nt < 16; ++nt) {
        float bias = b1[nt * 16 + n];
        v8f c; for (int r = 0; r < 8; ++r) c[r] = bias;
        #pragma unroll
        for (int kt = 0; kt < 2; ++kt) {
            Frag fb = wsB(W1F + (nt * 2 + kt) * 256, lane);
            c = wmma16(ax[kt], fb, c);
        }
        #pragma unroll
        for (int r = 0; r < 8; ++r)
            h1[(moff + r) * 256 + nt * 16 + n] = (_Float16)fmaxf(c[r], 0.0f);
    }
    asm volatile("s_wait_dscnt 0" ::: "memory");

    // -------- stage 2: z = h1 @ W2 + b2, M=16 K=256 N=128 ------------------
    for (int nt = 0; nt < 8; ++nt) {
        float bias = b2[nt * 16 + n];
        v8f c; for (int r = 0; r < 8; ++r) c[r] = bias;
        #pragma unroll
        for (int kt = 0; kt < 8; ++kt) {
            Frag fa = ldsA(h1, 256, kt, lane);
            Frag fb = wsB(W2F + (nt * 8 + kt) * 256, lane);
            c = wmma16(fa, fb, c);
        }
        #pragma unroll
        for (int r = 0; r < 8; ++r)
            zz[(moff + r) * 128 + nt * 16 + n] = (_Float16)c[r];
    }
    asm volatile("s_wait_dscnt 0" ::: "memory");

    // z A-fragments (K=128 -> 4 chunks), reused by 7 experts + gate.
    Frag az[4];
    #pragma unroll
    for (int kt = 0; kt < 4; ++kt) az[kt] = ldsA(zz, 128, kt, lane);

    // -------- stage 3: experts k=0..6 --------------------------------------
    // hk = relu(z @ Wr1[k] + br1[k]); pred_k = hk @ Wr2[k] + br2[k]
    for (int k = 0; k < 7; ++k) {
        float acc[8];
        #pragma unroll
        for (int r = 0; r < 8; ++r) acc[r] = 0.0f;
        const unsigned* WkF = Wr1F + k * 4096;
        for (int nt = 0; nt < 4; ++nt) {
            float bias = br1[k * 64 + nt * 16 + n];
            v8f c; for (int r = 0; r < 8; ++r) c[r] = bias;
            #pragma unroll
            for (int kt = 0; kt < 4; ++kt) {
                Frag fb = wsB(WkF + (nt * 4 + kt) * 256, lane);
                c = wmma16(az[kt], fb, c);
            }
            float wv2 = wr2[k * 64 + nt * 16 + n];       // Wr2[k][col]
            #pragma unroll
            for (int r = 0; r < 8; ++r) acc[r] += fmaxf(c[r], 0.0f) * wv2;
        }
        // reduce over the 16 columns held across the 16-lane half
        #pragma unroll
        for (int r = 0; r < 8; ++r) {
            float t = acc[r];
            t += __shfl_xor(t, 1);
            t += __shfl_xor(t, 2);
            t += __shfl_xor(t, 4);
            t += __shfl_xor(t, 8);
            acc[r] = t;
        }
        float bb = br2[k];
        if (n == 0)
            #pragma unroll
            for (int r = 0; r < 8; ++r) sh_pred[wv][moff + r][k] = acc[r] + bb;
    }
    asm volatile("s_wait_dscnt 0" ::: "memory");

    // -------- stage 4: gate logits, softmax, outputs -----------------------
    {
        float bias = (n < 7) ? bg[n] : 0.0f;
        v8f c; for (int r = 0; r < 8; ++r) c[r] = bias;
        #pragma unroll
        for (int kt = 0; kt < 4; ++kt) {
            Frag fb = wsB(WgF + kt * 256, lane);        // cols 7..15 zero-padded
            c = wmma16(az[kt], fb, c);
        }
        float w8[8], ys[8];
        #pragma unroll
        for (int r = 0; r < 8; ++r) {
            float v = (n < 7) ? c[r] : -3.0e38f;
            float mx = v;
            mx = fmaxf(mx, __shfl_xor(mx, 1));
            mx = fmaxf(mx, __shfl_xor(mx, 2));
            mx = fmaxf(mx, __shfl_xor(mx, 4));
            mx = fmaxf(mx, __shfl_xor(mx, 8));
            float e = __expf(v - mx);                   // 0 for padded lanes
            float s = e;
            s += __shfl_xor(s, 1);
            s += __shfl_xor(s, 2);
            s += __shfl_xor(s, 4);
            s += __shfl_xor(s, 8);
            float wgt = e / s;
            w8[r] = wgt;
            float t = (n < 7) ? wgt * sh_pred[wv][moff + r][n] : 0.0f;
            t += __shfl_xor(t, 1);
            t += __shfl_xor(t, 2);
            t += __shfl_xor(t, 4);
            t += __shfl_xor(t, 8);
            ys[r] = t;
        }
        if (n == 0)
            #pragma unroll
            for (int r = 0; r < 8; ++r) outSoft[base + moff + r] = ys[r];
        if (n < 7)
            #pragma unroll
            for (int r = 0; r < 8; ++r) outW[(base + moff + r) * 7 + n] = w8[r];
        if (lane < 16) {
            long row = base + lane;
            int idx = domain[row] - 1;
            idx = idx < 0 ? 0 : (idx > 6 ? 6 : idx);
            outHard[row] = sh_pred[wv][lane][idx];
        }
    }
}

extern "C" void kernel_launch(void* const* d_in, const int* in_sizes, int n_in,
                              void* d_out, int out_size, void* d_ws, size_t ws_size,
                              hipStream_t stream) {
    const float* x   = (const float*)d_in[0];
    const int*   dom = (const int*)d_in[1];
    const float* W1  = (const float*)d_in[2];
    const float* b1  = (const float*)d_in[3];
    const float* W2  = (const float*)d_in[4];
    const float* b2  = (const float*)d_in[5];
    const float* Wr1 = (const float*)d_in[6];
    const float* br1 = (const float*)d_in[7];
    const float* Wr2 = (const float*)d_in[8];
    const float* br2 = (const float*)d_in[9];
    const float* Wg  = (const float*)d_in[10];
    const float* bg  = (const float*)d_in[11];

    const int Bn = in_sizes[1];                 // 262144 rows

    // Workspace layout (dword offsets):
    //   W1F  @ 0      : 2 kt * 16 nt * 256 =  8192
    //   W2F  @ 8192   : 8 kt *  8 nt * 256 = 16384
    //   Wr1F @ 24576  : 7 * (4*4*256)      = 28672
    //   WgF  @ 53248  : 4 kt *  1 nt * 256 =  1024   (N=7 padded to 16)
    unsigned* ws   = (unsigned*)d_ws;
    unsigned* W1F  = ws;
    unsigned* W2F  = ws + 8192;
    unsigned* Wr1F = ws + 24576;
    unsigned* WgF  = ws + 53248;

    pack_b<<<(8192  + 255) / 256, 256, 0, stream>>>(W1,  W1F,  64,  256, 256, 0,    0,    8192);
    pack_b<<<(16384 + 255) / 256, 256, 0, stream>>>(W2,  W2F,  256, 128, 128, 0,    0,    16384);
    pack_b<<<(28672 + 255) / 256, 256, 0, stream>>>(Wr1, Wr1F, 128, 64,  64,  8192, 4096, 28672);
    pack_b<<<(1024  + 255) / 256, 256, 0, stream>>>(Wg,  WgF,  128, 7,   16,  0,    0,    1024);

    float* outHard = (float*)d_out;
    float* outSoft = outHard + Bn;
    float* outW    = outHard + 2L * Bn;

    moe_fused<<<Bn / 64, 128, 0, stream>>>(x, dom, b1, b2, br1, Wr2, br2, bg,
                                           W1F, W2F, Wr1F, WgF,
                                           outHard, outSoft, outW);
}
